// AtomicConvolution_70497593197149
// MI455X (gfx1250) — matrix operand order repaired and checked by reading, not compiled, not run on hardware
//
#include <hip/hip_runtime.h>

// CDNA5 / gfx1250 — wave32, WMMA 16x16x32 f16->f32
typedef __attribute__((ext_vector_type(16))) _Float16 v16h;
typedef __attribute__((ext_vector_type(8)))  float    v8f;

#define B_     16
#define N_     2048
#define M_     32
#define P_     48
#define T_     5
#define C_     (P_ * T_)          // 240 channels
#define WPB    8                  // waves per block (256 threads)
#define LOG2E  1.4426950408889634f

// ---------------------------------------------------------------------------
// Kernel 1: one wave per atom (b,n).
//   lane m loads neighbor m -> R_m, z_m into LDS.
//   B-matrix (32x16 f16): col t = onehot(z_K == type_t), half h of v16h ->
//     K = (h<8 ? h : h+8) + 8*(lane>=16)  (same order as A).
//   A-matrix (16x32 f16): row r = lane%16 -> p = 16*pi + r, half h ->
//     K = (h<8 ? h : h+8) + 8*(lane>=16)  (ISA 16-bit A 16x32 layout).
//   D (16x16 f32): vgpr v, lane -> row = v + 8*(lane>=16), col = lane%16.
//   rsf math uses raw v_exp_f32 / v_cos_f32 with pre-folded scale constants:
//     exp(-re*dR^2)   = exp2( dR^2 * (-re*log2(e)) )
//     cos(pi*R/rc)    = v_cos( R * (0.5/rc) )        (v_cos is revolutions)
//   Store unnormalized layer[b,n,c], c = t*48 + p  (two float4 per lane).
// ---------------------------------------------------------------------------
__global__ __launch_bounds__(256) void atomconv_rsf_wmma(
    const float* __restrict__ X,   // [B,N,3]
    const float* __restrict__ rc,  // [48]
    const float* __restrict__ rs,  // [48]
    const float* __restrict__ re,  // [48]
    const int*   __restrict__ Nbrs,   // [B,N,32]
    const int*   __restrict__ NbrsZ,  // [B,N,32]
    float*       __restrict__ out)    // [B,N,240] (unnormalized here)
{
    __shared__ float sR[WPB][M_];
    __shared__ int   sZ[WPB][M_];
    __shared__ float sRc[P_], sRs[P_], sNre2[P_], sHrc[P_];

    const int tid = threadIdx.x;
    if (tid < P_) {
        const float rcp = rc[tid];
        sRc[tid]   = rcp;
        sRs[tid]   = rs[tid];
        sNre2[tid] = -re[tid] * LOG2E;   // exp -> exp2 scale folded in
        sHrc[tid]  = 0.5f / rcp;         // cos(pi*R/rc) -> v_cos(R*0.5/rc)
    }

    const int lane = tid & 31;
    const int w    = tid >> 5;
    const int atom = blockIdx.x * WPB + w;   // 0..32767, grid sized exactly
    const int b    = atom >> 11;             // / 2048
    const int n    = atom & (N_ - 1);

    // --- neighbor gather: lane m handles neighbor m ---
    const size_t nbidx = (size_t)atom * M_ + lane;
    const int nbr = Nbrs[nbidx];
    const int zm  = NbrsZ[nbidx];

    const size_t xb = ((size_t)b * N_ + n)   * 3;
    const size_t nb = ((size_t)b * N_ + nbr) * 3;
    const float dx = X[nb + 0] - X[xb + 0];
    const float dy = X[nb + 1] - X[xb + 1];
    const float dz = X[nb + 2] - X[xb + 2];
    sR[w][lane] = sqrtf(dx * dx + dy * dy + dz * dz);
    sZ[w][lane] = zm;
    __syncthreads();

    // --- B matrix: one-hot atom-type mask, shared by all three p-tiles ---
    const int t = lane & 15;
    const int myType = (t == 0) ? 1 : (t == 1) ? 6 : (t == 2) ? 7
                     : (t == 3) ? 8 : (t == 4) ? 16 : -1;
    const int koffB = (lane & 16) >> 1;      // 0 or 8
    v16h bm;
#pragma unroll
    for (int h = 0; h < 16; ++h) {
        const int m = ((h < 8) ? h : h + 8) + koffB;
        bm[h] = (sZ[w][m] == myType) ? (_Float16)1.0f : (_Float16)0.0f;
    }

    // --- A-row R values: the K->m mapping is tile-independent, load once ---
    const int koffA = (lane & 16) >> 1;      // 0 or 8
    float Rv[16];
#pragma unroll
    for (int h = 0; h < 16; ++h) {
        const int m = ((h < 8) ? h : h + 8) + koffA;
        Rv[h] = sR[w][m];
    }

    // --- three 16x16x32 WMMAs cover p = 0..47 ---
    const int r = lane & 15;
    const size_t obase = ((size_t)b * N_ + n) * C_;

#pragma unroll
    for (int pi = 0; pi < 3; ++pi) {
        const int   p    = pi * 16 + r;
        const float rcp  = sRc[p];
        const float rsp  = sRs[p];
        const float nre2 = sNre2[p];
        const float hrc  = sHrc[p];
        v16h am;
#pragma unroll
        for (int h = 0; h < 16; ++h) {
            const float R    = Rv[h];
            const float dR   = R - rsp;
            const float g    = __builtin_amdgcn_exp2f((dR * dR) * nre2);
            const float cosr = __builtin_amdgcn_cosf(R * hrc);
            const float fcv  = __fmaf_rn(0.5f, cosr, 0.5f);
            const float fc   = (R <= rcp) ? fcv : 0.0f;   // branchless cndmask
            am[h] = (_Float16)(g * fc);
        }
        v8f acc = {};
        acc = __builtin_amdgcn_wmma_f32_16x16x32_f16(
            /*neg_a=*/false, am, /*neg_b=*/false, bm,
            /*c_mod=*/(short)0, acc, /*reuse_a=*/false, /*reuse_b=*/false);

        if (t < T_) {
            const int rowbase = (lane & 16) ? 8 : 0;
            const int c0 = t * P_ + pi * 16 + rowbase;   // 16B aligned
            float4 lo = make_float4(acc[0], acc[1], acc[2], acc[3]);
            float4 hi = make_float4(acc[4], acc[5], acc[6], acc[7]);
            *(float4*)(out + obase + c0)     = lo;
            *(float4*)(out + obase + c0 + 4) = hi;
        }
    }
}

// ---------------------------------------------------------------------------
// Kernel 2: BatchNorm1d(num_features=N), training-mode stats.
// One block per channel n: reduce 16*240 = 3840 values over (b,c),
// normalize in place: (v - mean) * rsqrt(var + 1e-5).
// ---------------------------------------------------------------------------
__global__ __launch_bounds__(256) void atomconv_bn(float* __restrict__ out)
{
    const int n   = blockIdx.x;
    const int tid = threadIdx.x;

    __shared__ float ssum[256], ssq[256];
    __shared__ float smean, sinv;

    float vals[15];
    float s = 0.0f, q = 0.0f;
#pragma unroll
    for (int k = 0; k < 15; ++k) {
        const int j = tid + k * 256;        // 0..3839
        const int b = j / C_;
        const int c = j - b * C_;
        const float v = out[((size_t)b * N_ + n) * C_ + c];
        vals[k] = v; s += v; q += v * v;
    }
    ssum[tid] = s; ssq[tid] = q;
    __syncthreads();
#pragma unroll
    for (int st = 128; st > 0; st >>= 1) {
        if (tid < st) { ssum[tid] += ssum[tid + st]; ssq[tid] += ssq[tid + st]; }
        __syncthreads();
    }
    if (tid == 0) {
        const float inv_n = 1.0f / (float)(B_ * C_);
        const float mean  = ssum[0] * inv_n;
        const float var   = ssq[0] * inv_n - mean * mean;
        smean = mean;
        sinv  = rsqrtf(var + 1e-5f);
    }
    __syncthreads();
    const float mean = smean, inv = sinv;
#pragma unroll
    for (int k = 0; k < 15; ++k) {
        const int j = tid + k * 256;
        const int b = j / C_;
        const int c = j - b * C_;
        out[((size_t)b * N_ + n) * C_ + c] = (vals[k] - mean) * inv;
    }
}

// ---------------------------------------------------------------------------
extern "C" void kernel_launch(void* const* d_in, const int* in_sizes, int n_in,
                              void* d_out, int out_size, void* d_ws, size_t ws_size,
                              hipStream_t stream)
{
    (void)in_sizes; (void)n_in; (void)out_size; (void)d_ws; (void)ws_size;
    const float* X    = (const float*)d_in[0];
    const float* rc   = (const float*)d_in[1];
    const float* rs   = (const float*)d_in[2];
    const float* re   = (const float*)d_in[3];
    const int*   Nbrs = (const int*)d_in[4];
    const int*   NbrZ = (const int*)d_in[5];
    float*       out  = (float*)d_out;

    // 32768 atoms, one wave each, 8 waves per 256-thread block -> 4096 blocks
    atomconv_rsf_wmma<<<(B_ * N_) / WPB, 256, 0, stream>>>(
        X, rc, rs, re, Nbrs, NbrZ, out);
    // one block per channel n
    atomconv_bn<<<N_, 256, 0, stream>>>(out);
}
